// GlycanGAT_88201448391393
// MI455X (gfx1250) — matrix operand order repaired and checked by reading, not compiled, not run on hardware
//
#include <hip/hip_runtime.h>

// GlycanGAT forward for MI455X (gfx1250): bf16 WMMA everywhere, fp32 accum.
// B=16, L=1024, D=256, H=8 (hd=32), NL=3, NU=1024.
// Round 3: break false register dependences — load all fragments of a K-step
// into distinct arrays before the WMMA burst (GEMMs), hoist V loads above the
// score WMMAs (attention) so each iteration is one load-clause + one wmma burst.

#define GB 16
#define GL 1024
#define GD 256
#define GH 8
#define GNL 3
#define GM (GB * GL)   // 16384 rows

typedef __attribute__((ext_vector_type(16))) __bf16 v16bf;
typedef __attribute__((ext_vector_type(8)))  float  v8f;

union Frag {
  int4     q[2];
  unsigned u[8];
  v16bf    v;
};

__device__ __forceinline__ unsigned short f2bf(float f) {
  unsigned u = __float_as_uint(f);
  u += 0x7FFFu + ((u >> 16) & 1u);        // round-to-nearest-even
  return (unsigned short)(u >> 16);
}
__device__ __forceinline__ unsigned packbf(float lo, float hi) {
  return (unsigned)f2bf(lo) | ((unsigned)f2bf(hi) << 16);
}

// ---------------- elementwise fp32 -> bf16 (weights) ----------------
__global__ __launch_bounds__(256) void k_f2bf(const float* __restrict__ a,
                                              unsigned short* __restrict__ b, int n) {
  for (int i = blockIdx.x * 256 + threadIdx.x; i < n; i += gridDim.x * 256)
    b[i] = f2bf(a[i]);
}

// ---------------- embedding gather ----------------
__global__ __launch_bounds__(256) void k_embed(const int* __restrict__ ut,
                                               const float* __restrict__ emb,
                                               float* __restrict__ xf,
                                               unsigned short* __restrict__ xb) {
  int row = blockIdx.x;           // 0..M-1
  int t   = threadIdx.x;          // 0..255
  int u   = ut[row];
  float v = emb[u * GD + t];
  xf[row * GD + t] = v;
  xb[row * GD + t] = f2bf(v);
}

// ---------------- shared 64x32 register-blocked mainloop ----------------
// acc[s*2+t] for 4 m-subtiles x 2 n-subtiles; all 6 fragments of a K-step are
// loaded into distinct registers before the 8-WMMA burst.
__device__ __forceinline__ void gemm_64x32(const unsigned short* const arow[4],
                                           const unsigned short* const wrow[2],
                                           int aoff, int boff, v8f acc[8]) {
#pragma unroll
  for (int k = 0; k < GD; k += 32) {
    Frag fb[2], fa[4];
#pragma unroll
    for (int t = 0; t < 2; ++t) {
      fb[t].q[0] = *(const int4*)(wrow[t] + k + boff);
      fb[t].q[1] = *(const int4*)(wrow[t] + k + boff + 16);
    }
#pragma unroll
    for (int s = 0; s < 4; ++s) {
      fa[s].q[0] = *(const int4*)(arow[s] + k + aoff);
      fa[s].q[1] = *(const int4*)(arow[s] + k + aoff + 16);
    }
    __builtin_prefetch(arow[0] + k + 32, 0, 1);
#pragma unroll
    for (int s = 0; s < 4; ++s) {
      acc[s * 2 + 0] = __builtin_amdgcn_wmma_f32_16x16x32_bf16(
          false, fa[s].v, false, fb[0].v, (short)0, acc[s * 2 + 0], false, false);
      acc[s * 2 + 1] = __builtin_amdgcn_wmma_f32_16x16x32_bf16(
          false, fa[s].v, false, fb[1].v, (short)0, acc[s * 2 + 1], false, false);
    }
  }
}

// ---------------- QKV GEMM: C[M,768] = Xbf[M,256] @ Wbf[768,256]^T + bias ------
// Q,K (cols 0..511) -> qk row-major bf16 [M,512].
// V   (cols 512..767) -> vt[(b*8+h)*32+d][L] bf16 (keys contiguous).
__global__ __launch_bounds__(256) void k_gemm_qkv(const unsigned short* __restrict__ A,
                                                  const unsigned short* __restrict__ W,
                                                  const float* __restrict__ bias,
                                                  unsigned short* __restrict__ qk,
                                                  unsigned short* __restrict__ vt) {
  const int lane = threadIdx.x & 31;
  const int wave = threadIdx.x >> 5;
  const int tile = blockIdx.x * 8 + wave;      // (M/64)*(768/32) = 6144 tiles
  const int mt = tile / 24, nt = tile % 24;    // mt: 0..255, nt: 0..23
  const bool hi  = lane >= 16;
  const int  l15 = lane & 15;
  const int aoff = hi ? 8 : 0;
  const int boff = hi ? 8 : 0;
  const unsigned short* arow[4];
  const unsigned short* wrow[2];
#pragma unroll
  for (int s = 0; s < 4; ++s) arow[s] = A + (size_t)(mt * 64 + s * 16 + l15) * GD;
#pragma unroll
  for (int t = 0; t < 2; ++t) wrow[t] = W + (size_t)(nt * 32 + t * 16 + l15) * GD;

  v8f acc[8] = {};
  gemm_64x32(arow, wrow, aoff, boff, acc);

#pragma unroll
  for (int t = 0; t < 2; ++t) {
    const int e  = nt * 32 + t * 16 + l15;     // output column
    const float bv = bias[e];
#pragma unroll
    for (int s = 0; s < 4; ++s) {
      const v8f& ac = acc[s * 2 + t];
      if (nt < 16) {                           // Q or K -> row-major [M,512]
#pragma unroll
        for (int r = 0; r < 8; ++r) {
          int m = mt * 64 + s * 16 + r + (hi ? 8 : 0);
          qk[(size_t)m * 512 + e] = f2bf(ac[r] + bv);
        }
      } else {                                 // V -> transposed (keys contiguous)
        int ev = e - 512;                      // h*32 + d
        int b  = (mt * 64 + s * 16) / GL;
        int l0 = (mt * 64 + s * 16) % GL + (hi ? 8 : 0);
        unsigned us[4];
#pragma unroll
        for (int r = 0; r < 4; ++r)
          us[r] = packbf(ac[2 * r] + bv, ac[2 * r + 1] + bv);
        *(int4*)(vt + ((size_t)(b * GH + (ev >> 5)) * 32 + (ev & 31)) * GL + l0) =
            *(int4*)us;
      }
    }
  }
}

// ---------------- online softmax step for one 16-query tile, 32 keys ----------
__device__ __forceinline__ void sm_step(const v8f& s0, const v8f& s1, int kb, int len,
                                        bool hi, float& mrun, float& lrun,
                                        v8f& accA, v8f& accB, Frag& fp) {
  const float scale = 0.1767766953f;           // 1/sqrt(32)
  const int k0 = kb + (hi ? 8 : 0);
  float sv[16];
  float tmax = -3.0e38f;
#pragma unroll
  for (int r = 0; r < 8; ++r) {
    float a = s0[r] * scale; if (k0 + r      >= len) a = -1.0e9f;
    float c = s1[r] * scale; if (k0 + 16 + r >= len) c = -1.0e9f;
    sv[r] = a; sv[8 + r] = c;
    tmax = fmaxf(tmax, fmaxf(a, c));
  }
  tmax = fmaxf(tmax, __shfl_xor(tmax, 16, 32));        // full 32-key row max
  const float mnew  = fmaxf(mrun, tmax);
  const float alpha = __expf(mrun - mnew);
  float p[16], lsum = 0.0f;
#pragma unroll
  for (int r = 0; r < 16; ++r) { p[r] = __expf(sv[r] - mnew); lsum += p[r]; }
  lsum += __shfl_xor(lsum, 16, 32);
  lrun = lrun * alpha + lsum;
  mrun = mnew;
#pragma unroll
  for (int r = 0; r < 8; ++r) {                        // rescale O accumulators
    float ar = __shfl(alpha, r + (hi ? 8 : 0), 32);
    accA[r] *= ar; accB[r] *= ar;
  }
#pragma unroll
  for (int j = 0; j < 4; ++j) fp.u[j]     = packbf(p[2 * j],     p[2 * j + 1]);
#pragma unroll
  for (int j = 0; j < 4; ++j) fp.u[4 + j] = packbf(p[8 + 2 * j], p[9 + 2 * j]);
}

// ---------------- attention: flash-style, one wave per (b,h,32-query group) ---
__global__ __launch_bounds__(256) void k_attn(const unsigned short* __restrict__ qk,
                                              const unsigned short* __restrict__ vt,
                                              const int* __restrict__ lengths,
                                              unsigned short* __restrict__ aob) {
  const int lane = threadIdx.x & 31;
  const int wave = threadIdx.x >> 5;
  const int tile = blockIdx.x * 8 + wave;      // B*H*(L/32) = 4096 tiles
  const int qg = tile & 31;                    // 32-query group
  const int h  = (tile >> 5) & 7;
  const int b  = tile >> 8;
  const bool hi  = lane >= 16;
  const int  l15 = lane & 15;
  const int  len = lengths[b];

  // persistent Q B-fragments (2 query tiles): B[d,q] = Q[q,d]; lane=q, contig d
  Frag fq0, fq1;
  {
    const unsigned short* q0 =
        qk + (size_t)(b * GL + qg * 32 + l15) * 512 + h * 32 + (hi ? 16 : 0);
    fq0.q[0] = *(const int4*)(q0);
    fq0.q[1] = *(const int4*)(q0 + 8);
    const unsigned short* q1 = q0 + (size_t)16 * 512;
    fq1.q[0] = *(const int4*)(q1);
    fq1.q[1] = *(const int4*)(q1 + 8);
  }

  const unsigned short* kbase = qk + (size_t)(b * GL) * 512 + GD + h * 32;
  const unsigned short* vbase = vt + (size_t)(b * GH + h) * 32 * GL;

  v8f acc00 = {}, acc01 = {}, acc10 = {}, acc11 = {};
  float mrun0 = -3.0e38f, lrun0 = 0.0f;
  float mrun1 = -3.0e38f, lrun1 = 0.0f;
  const int aoffd = hi ? 16 : 0;

  for (int kb = 0; kb < GL; kb += 32) {
    // one 8-load burst: K A-fragments + V B-fragments for 32 keys
    Frag fk0, fk1, fv0, fv1;
    const unsigned short* kr0 = kbase + (size_t)(kb + l15) * 512;
    fk0.q[0] = *(const int4*)(kr0 + aoffd);
    fk0.q[1] = *(const int4*)(kr0 + aoffd + 8);
    const unsigned short* kr1 = kbase + (size_t)(kb + 16 + l15) * 512;
    fk1.q[0] = *(const int4*)(kr1 + aoffd);
    fk1.q[1] = *(const int4*)(kr1 + aoffd + 8);
    const int kv = kb + (hi ? 8 : 0);
    const unsigned short* vr0 = vbase + (size_t)l15 * GL + kv;
    fv0.q[0] = *(const int4*)(vr0);
    fv0.q[1] = *(const int4*)(vr0 + 16);
    const unsigned short* vr1 = vbase + (size_t)(16 + l15) * GL + kv;
    fv1.q[0] = *(const int4*)(vr1);
    fv1.q[1] = *(const int4*)(vr1 + 16);

    // S^T = K_tile(16x32d) * Q^T(32d x 16q): 4 score wmmas
    v8f s00 = {}, s01 = {}, s10 = {}, s11 = {};
    s00 = __builtin_amdgcn_wmma_f32_16x16x32_bf16(false, fk0.v, false, fq0.v,
                                                  (short)0, s00, false, false);
    s01 = __builtin_amdgcn_wmma_f32_16x16x32_bf16(false, fk1.v, false, fq0.v,
                                                  (short)0, s01, false, false);
    s10 = __builtin_amdgcn_wmma_f32_16x16x32_bf16(false, fk0.v, false, fq1.v,
                                                  (short)0, s10, false, false);
    s11 = __builtin_amdgcn_wmma_f32_16x16x32_bf16(false, fk1.v, false, fq1.v,
                                                  (short)0, s11, false, false);

    Frag fp0, fp1;
    sm_step(s00, s01, kb, len, hi, mrun0, lrun0, acc00, acc01, fp0);
    sm_step(s10, s11, kb, len, hi, mrun1, lrun1, acc10, acc11, fp1);

    acc00 = __builtin_amdgcn_wmma_f32_16x16x32_bf16(false, fp0.v, false, fv0.v,
                                                    (short)0, acc00, false, false);
    acc01 = __builtin_amdgcn_wmma_f32_16x16x32_bf16(false, fp0.v, false, fv1.v,
                                                    (short)0, acc01, false, false);
    acc10 = __builtin_amdgcn_wmma_f32_16x16x32_bf16(false, fp1.v, false, fv0.v,
                                                    (short)0, acc10, false, false);
    acc11 = __builtin_amdgcn_wmma_f32_16x16x32_bf16(false, fp1.v, false, fv1.v,
                                                    (short)0, acc11, false, false);
  }

  // normalize and store bf16 into ao[M,256] at columns h*32 + d
  const float linv0 = 1.0f / lrun0;
  const float linv1 = 1.0f / lrun1;
#pragma unroll
  for (int r = 0; r < 8; ++r) {
    int srcl = r + (hi ? 8 : 0);
    float li0 = __shfl(linv0, srcl, 32);
    float li1 = __shfl(linv1, srcl, 32);
    int d0 = h * 32 + l15;
    size_t m0 = (size_t)(b * GL + qg * 32 + r + (hi ? 8 : 0));
    aob[m0 * GD + d0]      = f2bf(acc00[r] * li0);
    aob[m0 * GD + d0 + 16] = f2bf(acc01[r] * li0);
    size_t m1 = m0 + 16;
    aob[m1 * GD + d0]      = f2bf(acc10[r] * li1);
    aob[m1 * GD + d0 + 16] = f2bf(acc11[r] * li1);
  }
}

// ---------------- FF GEMM: C[M,256] = A @ W^T + bias (+residual)(+relu) -------
__global__ __launch_bounds__(256) void k_gemm_ff(const unsigned short* __restrict__ A,
                                                 const unsigned short* __restrict__ W,
                                                 const float* __restrict__ bias,
                                                 const float* __restrict__ residual,
                                                 float* __restrict__ outF,
                                                 unsigned short* __restrict__ outB,
                                                 int relu) {
  const int lane = threadIdx.x & 31;
  const int wave = threadIdx.x >> 5;
  const int tile = blockIdx.x * 8 + wave;      // (M/64)*(256/32) = 2048 tiles
  const int mt = tile >> 3, nt = tile & 7;
  const bool hi  = lane >= 16;
  const int  l15 = lane & 15;
  const int aoff = hi ? 8 : 0;
  const int boff = hi ? 8 : 0;
  const unsigned short* arow[4];
  const unsigned short* wrow[2];
#pragma unroll
  for (int s = 0; s < 4; ++s) arow[s] = A + (size_t)(mt * 64 + s * 16 + l15) * GD;
#pragma unroll
  for (int t = 0; t < 2; ++t) wrow[t] = W + (size_t)(nt * 32 + t * 16 + l15) * GD;

  v8f acc[8] = {};
  gemm_64x32(arow, wrow, aoff, boff, acc);

#pragma unroll
  for (int t = 0; t < 2; ++t) {
    const int col = nt * 32 + t * 16 + l15;
    const float bv = bias[col];
#pragma unroll
    for (int s = 0; s < 4; ++s) {
      const v8f& ac = acc[s * 2 + t];
#pragma unroll
      for (int r = 0; r < 8; ++r) {
        size_t m = (size_t)(mt * 64 + s * 16 + r + (hi ? 8 : 0));
        float v = ac[r] + bv;
        if (residual) v += residual[m * GD + col];
        if (relu) v = fmaxf(v, 0.0f);
        outF[m * GD + col] = v;
        outB[m * GD + col] = f2bf(v);
      }
    }
  }
}

// ---------------- readout ----------------
__global__ __launch_bounds__(256) void k_node(const float* __restrict__ xf,
                                              const int* __restrict__ lengths,
                                              float* __restrict__ out) {
  int tid = blockIdx.x * 256 + threadIdx.x;    // M*D elements
  int m = tid >> 8, c = tid & 255;
  int b = m >> 10, l = m & 1023;
  float v = xf[(size_t)m * GD + c];
  out[4096 + (size_t)m * GD + c] = (l < lengths[b]) ? v : 0.0f;
}

__global__ __launch_bounds__(256) void k_gpart(const float* __restrict__ out_node,
                                               float* __restrict__ part) {
  int b = blockIdx.x, j = blockIdx.y, t = threadIdx.x;
  float s = 0.0f;
  for (int l = j * 128; l < j * 128 + 128; ++l)
    s += out_node[(size_t)(b * GL + l) * GD + t];
  part[(size_t)(b * 8 + j) * GD + t] = s;
}

__global__ __launch_bounds__(256) void k_gfinal(const float* __restrict__ part,
                                                const int* __restrict__ lengths,
                                                float* __restrict__ out) {
  int b = blockIdx.x, t = threadIdx.x;
  float s = 0.0f;
#pragma unroll
  for (int j = 0; j < 8; ++j) s += part[(size_t)(b * 8 + j) * GD + t];
  int cnt = lengths[b] > 0 ? lengths[b] : 1;
  out[b * GD + t] = s / (float)cnt;
}

// ---------------- host ----------------
extern "C" void kernel_launch(void* const* d_in, const int* in_sizes, int n_in,
                              void* d_out, int out_size, void* d_ws, size_t ws_size,
                              hipStream_t stream) {
  (void)in_sizes; (void)n_in; (void)out_size; (void)ws_size;
  const int*   ut      = (const int*)d_in[0];
  const int*   lengths = (const int*)d_in[1];
  const float* emb     = (const float*)d_in[2];
  const float* wqkv    = (const float*)d_in[3];
  const float* bqkv    = (const float*)d_in[4];
  const float* wo      = (const float*)d_in[5];
  const float* bo      = (const float*)d_in[6];
  const float* wlin    = (const float*)d_in[7];
  const float* blin    = (const float*)d_in[8];
  float* out = (float*)d_out;

  char* ws = (char*)d_ws;
  size_t off = 0;
  auto alloc = [&](size_t bytes) -> void* {
    void* p = ws + off;
    off += (bytes + 255) & ~(size_t)255;
    return p;
  };
  unsigned short* xb    = (unsigned short*)alloc((size_t)GM * GD * 2);
  float*          xf    = (float*)alloc((size_t)GM * GD * 4);
  unsigned short* qkb   = (unsigned short*)alloc((size_t)GM * 2 * GD * 2);
  unsigned short* vtb   = (unsigned short*)alloc((size_t)GB * GH * 32 * GL * 2);
  unsigned short* aob   = (unsigned short*)alloc((size_t)GM * GD * 2);
  float*          hf    = (float*)alloc((size_t)GM * GD * 4);
  unsigned short* hb    = (unsigned short*)alloc((size_t)GM * GD * 2);
  unsigned short* wqkvb = (unsigned short*)alloc((size_t)GNL * 3 * GD * GD * 2);
  unsigned short* wob   = (unsigned short*)alloc((size_t)GNL * GD * GD * 2);
  unsigned short* wlinb = (unsigned short*)alloc((size_t)GNL * GD * GD * 2);
  float*          gpart = (float*)alloc((size_t)GB * 8 * GD * 4);

  k_f2bf<<<1024, 256, 0, stream>>>(wqkv, wqkvb, GNL * 3 * GD * GD);
  k_f2bf<<<1024, 256, 0, stream>>>(wo,   wob,   GNL * GD * GD);
  k_f2bf<<<1024, 256, 0, stream>>>(wlin, wlinb, GNL * GD * GD);
  k_embed<<<GM, 256, 0, stream>>>(ut, emb, xf, xb);

  for (int i = 0; i < GNL; ++i) {
    k_gemm_qkv<<<(GM / 64) * 24 / 8, 256, 0, stream>>>(
        xb, wqkvb + (size_t)i * 3 * GD * GD, bqkv + i * 3 * GD, qkb, vtb);
    k_attn<<<(GB * GH * (GL / 32)) / 8, 256, 0, stream>>>(qkb, vtb, lengths, aob);
    k_gemm_ff<<<(GM / 64) * 8 / 8, 256, 0, stream>>>(
        aob, wob + (size_t)i * GD * GD, bo + i * GD, xf, hf, hb, 0);
    k_gemm_ff<<<(GM / 64) * 8 / 8, 256, 0, stream>>>(
        hb, wlinb + (size_t)i * GD * GD, blin + i * GD, nullptr, xf, xb,
        (i < GNL - 1) ? 1 : 0);
  }

  k_node<<<(GM * GD) / 256, 256, 0, stream>>>(xf, lengths, out);
  k_gpart<<<dim3(GB, 8), 256, 0, stream>>>(out + 4096, gpart);
  k_gfinal<<<GB, 256, 0, stream>>>(gpart, lengths, out);
}
